// CausalSelfAttention_56848187130045
// MI455X (gfx1250) — matrix operand order repaired and checked by reading, not compiled, not run on hardware
//
#include <hip/hip_runtime.h>

#define B_  4
#define T_  2048
#define H_  16
#define D_  64
#define HID_ 1024

typedef __attribute__((ext_vector_type(16))) __bf16 v16bf;
typedef __attribute__((ext_vector_type(8)))  __bf16 v8bf;
typedef __attribute__((ext_vector_type(8)))  float  v8f;

union FragBF { v16bf v; v8bf h[2]; };

__device__ __forceinline__ v8f wmma_bf16(v16bf a, v16bf b, v8f c) {
  return __builtin_amdgcn_wmma_f32_16x16x32_bf16(false, a, false, b, (short)0, c,
                                                 false, false);
}

// A-matrix fragment (16x32 bf16): lane<16 -> row=row0+lane, K={k..k+15}
//                                  lane>=16 -> row=row0+lane-16, K={k+16..k+31}
__device__ __forceinline__ v16bf load_a_frag(const __bf16* __restrict__ base, int ld,
                                             int row0, int k, int lane) {
  int half = lane >> 4, lr = lane & 15;
  const __bf16* p = base + (size_t)(row0 + lr) * ld + (k + half * 16);
  FragBF f;
  f.h[0] = *(const v8bf*)(p);
  f.h[1] = *(const v8bf*)(p + 8);
  return f.v;
}

// B-matrix fragment (32x16 bf16, same K order as A frag): lane holds column n=col0+(lane&15);
// half-wave selects contiguous K block of 16: K = k + (lane>>4)*16 .. +15.
__device__ __forceinline__ v16bf load_b_frag(const __bf16* __restrict__ base, int ld,
                                             int col0, int k, int lane) {
  int half = lane >> 4, lr = lane & 15;
  const __bf16* p = base + (size_t)(col0 + lr) * ld + (k + half * 16);
  FragBF f;
  f.h[0] = *(const v8bf*)(p);
  f.h[1] = *(const v8bf*)(p + 8);
  return f.v;
}

// B fragment with row/K clamping for causal tail tiles (clamped chunks are fully masked).
__device__ __forceinline__ v16bf load_b_frag_clamp(const __bf16* __restrict__ base, int ld,
                                                   int col0, int k, int lane,
                                                   int maxRow, int maxK) {
  int half = lane >> 4, lr = lane & 15;
  int row = col0 + lr; if (row > maxRow) row = maxRow;
  int k0 = k + half * 16; int k1 = k0 + 8;
  if (k0 > maxK) k0 = maxK;
  if (k1 > maxK) k1 = maxK;
  const __bf16* p = base + (size_t)row * ld;
  FragBF f;
  f.h[0] = *(const v8bf*)(p + k0);
  f.h[1] = *(const v8bf*)(p + k1);
  return f.v;
}

// ---------------------------------------------------------------- convert
__global__ void cvt_f32_bf16(const float* __restrict__ src, __bf16* __restrict__ dst, int n) {
  int i = blockIdx.x * blockDim.x + threadIdx.x;
  if (i < n) dst[i] = (__bf16)src[i];
}

// ---------------------------------------------------------------- GEMM  C[M,N] = A[M,K] * W[N,K]^T
// One wave per 32x64 output tile: 12 b128 loads issued as one clause, then 8 WMMA.
template <typename OutT>
__global__ __launch_bounds__(32, 1) void gemm_bf16_wmma(
    const __bf16* __restrict__ A, const __bf16* __restrict__ W,
    OutT* __restrict__ C, int M, int N, int K) {
  int ntiles = N >> 6;
  int tile = blockIdx.x;
  int m0 = (tile / ntiles) << 5;
  int n0 = (tile % ntiles) << 6;
  int lane = threadIdx.x;

  v8f acc[2][4] = {};
  for (int k = 0; k < K; k += 32) {
    v16bf a0 = load_a_frag(A, K, m0,      k, lane);
    v16bf a1 = load_a_frag(A, K, m0 + 16, k, lane);
    v16bf b[4];
#pragma unroll
    for (int j = 0; j < 4; ++j) b[j] = load_b_frag(W, K, n0 + 16 * j, k, lane);
    if (k + 64 < K) {
      __builtin_prefetch((const void*)(A + (size_t)(m0 + (lane & 15)) * K + k + 64), 0, 1);
      __builtin_prefetch((const void*)(W + (size_t)(n0 + (lane & 15)) * K + k + 64), 0, 1);
    }
#pragma unroll
    for (int j = 0; j < 4; ++j) {
      acc[0][j] = wmma_bf16(a0, b[j], acc[0][j]);
      acc[1][j] = wmma_bf16(a1, b[j], acc[1][j]);
    }
  }
  int half = lane >> 4, lr = lane & 15;
#pragma unroll
  for (int mi = 0; mi < 2; ++mi)
#pragma unroll
    for (int j = 0; j < 4; ++j)
#pragma unroll
      for (int r = 0; r < 8; ++r) {
        int row = m0 + mi * 16 + half * 8 + r;   // C layout: VGPR r -> M=r / M=8+r
        int col = n0 + j * 16 + lr;              // lane -> N
        C[(size_t)row * N + col] = (OutT)acc[mi][j][r];
      }
}

// ---------------------------------------------------------------- RoPE + scatter
__global__ void rope_scatter(const __bf16* __restrict__ Y,
                             const float* __restrict__ cosT, const float* __restrict__ sinT,
                             __bf16* __restrict__ Q, __bf16* __restrict__ Kd,
                             __bf16* __restrict__ Vt) {
  int idx = blockIdx.x * blockDim.x + threadIdx.x;     // B*T*H*32 threads
  int d = idx & 31;
  int h = (idx >> 5) & (H_ - 1);
  int t = (idx >> 9) & (T_ - 1);
  int b = idx >> 20;
  if (b >= B_) return;

  size_t yrow = (size_t)(b * T_ + t) * (3 * HID_);
  float c1 = cosT[t * D_ + d],      s1 = sinT[t * D_ + d];
  float c2 = cosT[t * D_ + d + 32], s2 = sinT[t * D_ + d + 32];
  int bh = b * H_ + h;
  size_t qk = ((size_t)bh * T_ + t) * D_;
  {
    float x1 = (float)Y[yrow + h * D_ + d];
    float x2 = (float)Y[yrow + h * D_ + d + 32];
    Q[qk + d]      = (__bf16)(x1 * c1 - x2 * s1);
    Q[qk + d + 32] = (__bf16)(x2 * c2 + x1 * s2);
  }
  {
    float x1 = (float)Y[yrow + HID_ + h * D_ + d];
    float x2 = (float)Y[yrow + HID_ + h * D_ + d + 32];
    Kd[qk + d]      = (__bf16)(x1 * c1 - x2 * s1);
    Kd[qk + d + 32] = (__bf16)(x2 * c2 + x1 * s2);
  }
  {
    Vt[((size_t)bh * D_ + d)      * T_ + t] = Y[yrow + 2 * HID_ + h * D_ + d];
    Vt[((size_t)bh * D_ + d + 32) * T_ + t] = Y[yrow + 2 * HID_ + h * D_ + d + 32];
  }
}

// ---------------------------------------------------------------- flash attention
// One wave per (b,h, 16-query tile). kv step = 32: 4 WMMA (S) + 1 WMMA (rowsum) + 4 WMMA (PV).
// K/V fragments loaded in pairs to bound register pressure (no spills).
__global__ __launch_bounds__(32, 1) void attn_fa(
    const __bf16* __restrict__ Q, const __bf16* __restrict__ Kb,
    const __bf16* __restrict__ Vt, __bf16* __restrict__ Ctx) {
  __shared__ __align__(16) __bf16 Pld[16 * 32];
  int q0 = blockIdx.x << 4;
  int bh = blockIdx.y;
  int b = bh >> 4, h = bh & (H_ - 1);
  int lane = threadIdx.x, half = lane >> 4, lr = lane & 15;

  const __bf16* Qm = Q  + (size_t)bh * T_ * D_;
  const __bf16* Km = Kb + (size_t)bh * T_ * D_;
  const __bf16* Vm = Vt + (size_t)bh * D_ * T_;

  v16bf aq0 = load_a_frag(Qm, D_, q0, 0,  lane);
  v16bf aq1 = load_a_frag(Qm, D_, q0, 32, lane);

  v16bf onesf;                                  // ones B-fragment for WMMA row sums
#pragma unroll
  for (int i = 0; i < 16; ++i) onesf[i] = (__bf16)1.0f;

  v8f o[4] = {};
  float mrow[8], lrow[8];
#pragma unroll
  for (int r = 0; r < 8; ++r) { mrow[r] = -1e30f; lrow[r] = 0.f; }

  for (int kv0 = 0; kv0 <= q0 + 15; kv0 += 32) {
    // ---- S = Q K^T: fragments in pairs (4 b128 per wait)
    v8f st0 = {}, st1 = {};
    {
      v16bf bk0 = load_b_frag_clamp(Km, D_, kv0, 0,  lane, T_ - 1, 1 << 28);
      v16bf bk1 = load_b_frag_clamp(Km, D_, kv0, 32, lane, T_ - 1, 1 << 28);
      st0 = wmma_bf16(aq0, bk0, st0);
      st0 = wmma_bf16(aq1, bk1, st0);
    }
    {
      v16bf bk2 = load_b_frag_clamp(Km, D_, kv0 + 16, 0,  lane, T_ - 1, 1 << 28);
      v16bf bk3 = load_b_frag_clamp(Km, D_, kv0 + 16, 32, lane, T_ - 1, 1 << 28);
      st1 = wmma_bf16(aq0, bk2, st1);
      st1 = wmma_bf16(aq1, bk3, st1);
    }

    // ---- scale + causal mask + per-lane partial max
    float s0a[8], s1a[8], mx[8];
#pragma unroll
    for (int r = 0; r < 8; ++r) {
      int qi = q0 + half * 8 + r;
      float s0 = st0[r] * 0.125f;   // 1/sqrt(64)
      float s1 = st1[r] * 0.125f;
      if (kv0 + lr > qi)      s0 = -1e30f;
      if (kv0 + 16 + lr > qi) s1 = -1e30f;
      s0a[r] = s0; s1a[r] = s1;
      mx[r] = fmaxf(s0, s1);
    }
    // ---- batched xor-reduction over the 16-lane half-groups
#pragma unroll
    for (int mask = 1; mask <= 8; mask <<= 1) {
      float t[8];
#pragma unroll
      for (int r = 0; r < 8; ++r) t[r] = __shfl_xor(mx[r], mask);
#pragma unroll
      for (int r = 0; r < 8; ++r) mx[r] = fmaxf(mx[r], t[r]);
    }

    // ---- online softmax state + P in bf16 to LDS
    float al[8];
#pragma unroll
    for (int r = 0; r < 8; ++r) {
      float mn = fmaxf(mrow[r], mx[r]);
      al[r] = __expf(mrow[r] - mn);
      mrow[r] = mn;
      float p0 = __expf(s0a[r] - mn);
      float p1 = __expf(s1a[r] - mn);
      Pld[(half * 8 + r) * 32 + lr]      = (__bf16)p0;
      Pld[(half * 8 + r) * 32 + 16 + lr] = (__bf16)p1;
#pragma unroll
      for (int dt = 0; dt < 4; ++dt) o[dt][r] *= al[r];
    }
    __syncthreads();
    FragBF pf;                                  // re-read P in A-fragment layout
    pf.h[0] = *(const v8bf*)&Pld[lr * 32 + half * 16];
    pf.h[1] = *(const v8bf*)&Pld[lr * 32 + half * 16 + 8];
    __syncthreads();

    // ---- row sums via WMMA: P @ ones replicates the row sum into every column
    v8f zero = {};
    v8f ls = wmma_bf16(pf.v, onesf, zero);
#pragma unroll
    for (int r = 0; r < 8; ++r) lrow[r] = lrow[r] * al[r] + ls[r];

    // ---- O += P V: fragments in pairs
    {
      v16bf bv0 = load_b_frag_clamp(Vm, T_, 0,  kv0, lane, 1 << 28, T_ - 8);
      v16bf bv1 = load_b_frag_clamp(Vm, T_, 16, kv0, lane, 1 << 28, T_ - 8);
      o[0] = wmma_bf16(pf.v, bv0, o[0]);
      o[1] = wmma_bf16(pf.v, bv1, o[1]);
    }
    {
      v16bf bv2 = load_b_frag_clamp(Vm, T_, 32, kv0, lane, 1 << 28, T_ - 8);
      v16bf bv3 = load_b_frag_clamp(Vm, T_, 48, kv0, lane, 1 << 28, T_ - 8);
      o[2] = wmma_bf16(pf.v, bv2, o[2]);
      o[3] = wmma_bf16(pf.v, bv3, o[3]);
    }
  }

#pragma unroll
  for (int dt = 0; dt < 4; ++dt)
#pragma unroll
    for (int r = 0; r < 8; ++r) {
      float val = o[dt][r] / lrow[r];
      size_t row = (size_t)(b * T_ + q0 + half * 8 + r);
      Ctx[row * HID_ + h * D_ + dt * 16 + lr] = (__bf16)val;
    }
}

// ---------------------------------------------------------------- launch
extern "C" void kernel_launch(void* const* d_in, const int* in_sizes, int n_in,
                              void* d_out, int out_size, void* d_ws, size_t ws_size,
                              hipStream_t stream) {
  (void)in_sizes; (void)n_in; (void)out_size; (void)ws_size;
  const float* x    = (const float*)d_in[0];
  const float* cosT = (const float*)d_in[1];
  const float* sinT = (const float*)d_in[2];
  const float* wq   = (const float*)d_in[3];
  const float* wk   = (const float*)d_in[4];
  const float* wv   = (const float*)d_in[5];
  const float* wo   = (const float*)d_in[6];
  float* out = (float*)d_out;

  char* ws = (char*)d_ws;
  // workspace layout (bytes)
  __bf16* Xb   = (__bf16*)(ws + 0);            // 8192x1024 bf16  = 16 MiB
  __bf16* Wqkv = (__bf16*)(ws + 16777216);     // 3072x1024 bf16  =  6 MiB
  __bf16* Wob  = (__bf16*)(ws + 23068672);     // 1024x1024 bf16  =  2 MiB
  __bf16* Y    = (__bf16*)(ws + 25165824);     // 8192x3072 bf16  = 48 MiB
  __bf16* Qb   = (__bf16*)(ws + 75497472);     // [B,H,T,D] bf16  = 16 MiB
  __bf16* Kbuf = (__bf16*)(ws + 92274688);     // [B,H,T,D] bf16  = 16 MiB
  __bf16* Vtb  = (__bf16*)(ws + 109051904);    // [B,H,D,T] bf16  = 16 MiB
  __bf16* Ctx  = Y;                            // reuse consumed QKV buffer

  const int nX = B_ * T_ * HID_;               // 8388608
  const int nW = HID_ * HID_;                  // 1048576
  cvt_f32_bf16<<<(nX + 255) / 256, 256, 0, stream>>>(x,  Xb,            nX);
  cvt_f32_bf16<<<(nW + 255) / 256, 256, 0, stream>>>(wq, Wqkv,          nW);
  cvt_f32_bf16<<<(nW + 255) / 256, 256, 0, stream>>>(wk, Wqkv + nW,     nW);
  cvt_f32_bf16<<<(nW + 255) / 256, 256, 0, stream>>>(wv, Wqkv + 2 * nW, nW);
  cvt_f32_bf16<<<(nW + 255) / 256, 256, 0, stream>>>(wo, Wob,           nW);

  // fused QKV projection: [8192,1024] x [3072,1024]^T -> [8192,3072]
  gemm_bf16_wmma<__bf16><<<(8192 / 32) * (3072 / 64), 32, 0, stream>>>(
      Xb, Wqkv, Y, 8192, 3072, 1024);

  const int nR = B_ * T_ * H_ * 32;            // 4194304
  rope_scatter<<<nR / 256, 256, 0, stream>>>(Y, cosT, sinT, Qb, Kbuf, Vtb);

  attn_fa<<<dim3(T_ / 16, B_ * H_), 32, 0, stream>>>(Qb, Kbuf, Vtb, Ctx);

  // output projection: [8192,1024] x [1024,1024]^T -> f32 out
  gemm_bf16_wmma<float><<<(8192 / 32) * (1024 / 64), 32, 0, stream>>>(
      Ctx, Wob, out, 8192, 1024, 1024);
}
